// CellSetAttentionNetwork_40114994545231
// MI455X (gfx1250) — matrix operand order repaired and checked by reading, not compiled, not run on hardware
//
#include <hip/hip_runtime.h>

typedef __attribute__((ext_vector_type(16))) _Float16 v16h;
typedef __attribute__((ext_vector_type(8)))  _Float16 v8h;
typedef __attribute__((ext_vector_type(8)))  float    v8f;
typedef unsigned int u32x4 __attribute__((ext_vector_type(4)));
typedef int          i32x4 __attribute__((ext_vector_type(4)));
typedef int          i32x8 __attribute__((ext_vector_type(8)));

#define DIM   512
#define NSEQ  2048
#define BATCH 8
#define KB    32   // keys staged per block-iteration

#if __has_builtin(__builtin_amdgcn_tensor_load_to_lds) && \
    __has_builtin(__builtin_amdgcn_s_wait_tensorcnt)
#define USE_TDM 1
#else
#define USE_TDM 0
#endif

static __device__ __forceinline__ v8f wmma_f16(v16h a, v16h b, v8f c) {
  // D = A(16x32 f16) * B(32x16 f16) + C(16x16 f32)
  return __builtin_amdgcn_wmma_f32_16x16x32_f16(false, a, false, b, (short)0, c,
                                                false, false);
}

static __device__ __forceinline__ v16h cat8(v8h lo, v8h hi) {
  return __builtin_shufflevector(lo, hi, 0, 1, 2, 3, 4, 5, 6, 7,
                                 8, 9, 10, 11, 12, 13, 14, 15);
}

static __device__ __forceinline__ v8f splat8(float x) {
  v8f v;
#pragma unroll
  for (int i = 0; i < 8; ++i) v[i] = x;
  return v;
}

#if USE_TDM
// Issue one TDM 2D tile load: tile_d1 rows of tile_d0 elements (f16),
// source row stride stride0 (elements), into contiguous LDS at ldsptr.
// Descriptor packing per CDNA5 ISA 8.3/8.4 (data_size=1 -> 2 bytes, type=2).
static __device__ __forceinline__ void tdm_load_2d(
    const void* gptr, void* ldsptr, unsigned tensor_d0, unsigned tensor_d1,
    unsigned stride0, unsigned tile_d0, unsigned tile_d1) {
  unsigned long long ga = (unsigned long long)(uintptr_t)gptr;
  unsigned lds = (unsigned)(uintptr_t)ldsptr;  // flat LDS addr: low 32 = offset
  u32x4 g0;
  g0[0] = 1u;                                              // count=1
  g0[1] = lds;                                             // lds_addr
  g0[2] = (unsigned)ga;                                    // global_addr[31:0]
  g0[3] = (unsigned)((ga >> 32) & 0x01FFFFFFu) | (2u << 30);  // addr[56:32]|type=2
  i32x8 g1;
  g1[0] = (int)(1u << 16);                                 // data_size=1 (2B)
  g1[1] = (int)((tensor_d0 & 0xFFFFu) << 16);              // tensor_dim0[15:0]
  g1[2] = (int)(((tensor_d0 >> 16) & 0xFFFFu) |
                ((tensor_d1 & 0xFFFFu) << 16));            // dim0 hi | dim1 lo
  g1[3] = (int)(((tensor_d1 >> 16) & 0xFFFFu) |
                ((tile_d0 & 0xFFFFu) << 16));              // dim1 hi | tile_dim0
  g1[4] = (int)(tile_d1 & 0xFFFFu);                        // tile_dim1 (dim2=0)
  g1[5] = (int)stride0;                                    // dim0_stride[31:0]
  g1[6] = 0;
  g1[7] = 0;
  i32x4 gz = {0, 0, 0, 0};
#if __clang_major__ >= 23
  i32x8 gz8 = {0, 0, 0, 0, 0, 0, 0, 0};
  __builtin_amdgcn_tensor_load_to_lds(g0, g1, gz, gz, gz8, 0);
#else
  __builtin_amdgcn_tensor_load_to_lds(g0, g1, gz, gz, 0);
#endif
}
#endif

// ---------------------------------------------------------------------------
// Kernel 0: transpose weights and convert to f16.  WT[h][d] = (f16)W[d][h]
// ---------------------------------------------------------------------------
__global__ __launch_bounds__(256) void wt_kernel(
    const float* __restrict__ Wq, const float* __restrict__ Wk,
    const float* __restrict__ Wv, const float* __restrict__ Wo,
    _Float16* __restrict__ WqT, _Float16* __restrict__ WkT,
    _Float16* __restrict__ WvT, _Float16* __restrict__ WoT) {
  int idx = blockIdx.x * blockDim.x + threadIdx.x;  // 512*512 threads
  int h = idx >> 9;
  int d = idx & 511;
  int src = d * DIM + h;
  WqT[idx] = (_Float16)Wq[src];
  WkT[idx] = (_Float16)Wk[src];
  WvT[idx] = (_Float16)Wv[src];
  WoT[idx] = (_Float16)Wo[src];
}

// ---------------------------------------------------------------------------
// Kernel 1: Q = XWq+bq, K = XWk+bk, V = XWv+bv  (f16 outputs; V transposed)
// One wave handles a 16-row tile with A-fragments resident in VGPRs.
// ---------------------------------------------------------------------------
__global__ __launch_bounds__(128) void qkv_kernel(
    const float* __restrict__ X, const _Float16* __restrict__ WT0,
    const _Float16* __restrict__ WT1, const _Float16* __restrict__ WT2,
    const float* __restrict__ b0, const float* __restrict__ b1,
    const float* __restrict__ b2, _Float16* __restrict__ Qh,
    _Float16* __restrict__ Kh, _Float16* __restrict__ Vt) {
  const int lane = threadIdx.x & 31;
  const int wv = threadIdx.x >> 5;
  const int tile = blockIdx.x * 4 + wv;  // 0..1023
  const int rowbase = tile * 16;
  const int m = lane & 15;
  const int hl = lane >> 4;

  // Resident A-fragments: rows of X converted to f16 (16 k-steps over D=512).
  v16h afr[16];
  {
    const float* xr = X + (size_t)(rowbase + m) * DIM + hl * 16;
#pragma unroll
    for (int t = 0; t < 16; ++t) {
      const float4* p0 = (const float4*)(xr + t * 32);
      const float4* p1 = (const float4*)(xr + t * 32 + 8);
      float4 a0 = p0[0], a1 = p0[1], b0v = p1[0], b1v = p1[1];
      v16h a;
      a[0] = (_Float16)a0.x;   a[1] = (_Float16)a0.y;
      a[2] = (_Float16)a0.z;   a[3] = (_Float16)a0.w;
      a[4] = (_Float16)a1.x;   a[5] = (_Float16)a1.y;
      a[6] = (_Float16)a1.z;   a[7] = (_Float16)a1.w;
      a[8] = (_Float16)b0v.x;  a[9] = (_Float16)b0v.y;
      a[10] = (_Float16)b0v.z; a[11] = (_Float16)b0v.w;
      a[12] = (_Float16)b1v.x; a[13] = (_Float16)b1v.y;
      a[14] = (_Float16)b1v.z; a[15] = (_Float16)b1v.w;
      afr[t] = a;
    }
  }

  const _Float16* WTs[3] = {WT0, WT1, WT2};
  const float* biases[3] = {b0, b1, b2};
  const int b = rowbase / NSEQ;
  const int nloc = rowbase % NSEQ;

  for (int mat = 0; mat < 3; ++mat) {
    const _Float16* WT = WTs[mat];
    const float* bias = biases[mat];
    for (int ht = 0; ht < 32; ++ht) {
      v8f acc = splat8(bias[ht * 16 + m]);
      const _Float16* wrow = WT + (size_t)(ht * 16 + m) * DIM + hl * 16;
#pragma unroll
      for (int t = 0; t < 16; ++t) {
        v16h bf = *(const v16h*)(wrow + t * 32);
        acc = wmma_f16(afr[t], bf, acc);
      }
      if (mat < 2) {
        _Float16* out = (mat == 0) ? Qh : Kh;
#pragma unroll
        for (int i = 0; i < 8; ++i)
          out[(size_t)(rowbase + i + 8 * hl) * DIM + ht * 16 + m] =
              (_Float16)acc[i];
      } else {
        v8h pk;
#pragma unroll
        for (int i = 0; i < 8; ++i) pk[i] = (_Float16)acc[i];
        *(v8h*)(Vt + (size_t)b * DIM * NSEQ + (size_t)(ht * 16 + m) * NSEQ +
                nloc + 8 * hl) = pk;
      }
    }
  }
}

// ---------------------------------------------------------------------------
// Kernel 2: attention.  Workgroup = 64 query rows (4 waves x 16 rows).
// K/V 32-key tiles staged into double-buffered LDS by the Tensor Data Mover
// (TENSORcnt-tracked, overlapped with WMMA); all 4 waves consume fragments
// from LDS.  exp without max-shift (scores O(1)); divide by row sum at end.
// ---------------------------------------------------------------------------
__global__ __launch_bounds__(128) void attn_kernel(
    const _Float16* __restrict__ Qh, const _Float16* __restrict__ Kh,
    const _Float16* __restrict__ Vt, _Float16* __restrict__ Ctx) {
  __shared__ _Float16 kbuf[2][KB * DIM];   // 2 x 32 KB: keys (row-major)
  __shared__ _Float16 vbuf[2][DIM * KB];   // 2 x 32 KB: V^T slice [h][key]
  __shared__ _Float16 pbuf[4][16 * 32];    // per-wave P re-striping

  const int tid = threadIdx.x;
  const int lane = tid & 31;
  const int wv = tid >> 5;
  const int rowblk = blockIdx.x * 64;
  const int rowbase = rowblk + wv * 16;
  const int b = rowblk / NSEQ;
  const int m = lane & 15;
  const int hl = lane >> 4;

  const _Float16* kbase = Kh + (size_t)b * NSEQ * DIM;
  const _Float16* vbase = Vt + (size_t)b * DIM * NSEQ;

  // Q fragments resident in VGPRs (A layout), straight from global.
  v16h qfr[16];
  {
    const _Float16* qr = Qh + (size_t)(rowbase + m) * DIM + hl * 16;
#pragma unroll
    for (int t = 0; t < 16; ++t) {
      v8h lo = *(const v8h*)(qr + t * 32);
      v8h hi = *(const v8h*)(qr + t * 32 + 8);
      qfr[t] = cat8(lo, hi);
    }
  }

  v8f ctx[32];
#pragma unroll
  for (int t = 0; t < 32; ++t) ctx[t] = splat8(0.f);
  float lsum[8];
#pragma unroll
  for (int i = 0; i < 8; ++i) lsum[i] = 0.f;

  const float kExpScale = 1.4426950408889634f * 0.044194173824159216f;

  // --- staging helpers (TDM primary, cooperative-copy fallback) ---
#define STAGE_TILES(bufi, jb)                                                \
  do {                                                                       \
    stage_k_v((bufi), (jb));                                                 \
  } while (0)

  auto stage_k_v = [&](int bufi, int jb) {
#if USE_TDM
    if (wv == 0) {
      // K tile: 32 contiguous rows of 512 f16.
      tdm_load_2d(kbase + (size_t)jb * DIM, kbuf[bufi], DIM, KB, DIM, DIM, KB);
      // V^T tile: 512 rows of 32 f16, source row stride NSEQ.
      tdm_load_2d(vbase + jb, vbuf[bufi], KB, DIM, NSEQ, KB, DIM);
    }
#else
    {
      const v8h* src = (const v8h*)(kbase + (size_t)jb * DIM);
      v8h* dst = (v8h*)kbuf[bufi];
      for (int i = tid; i < KB * DIM / 8; i += 128) dst[i] = src[i];
    }
    {
      const v8h* src = (const v8h*)(vbase + jb);
      v8h* dst = (v8h*)vbuf[bufi];
      for (int i = tid; i < DIM * KB / 8; i += 128) {
        int h = i >> 2, c = i & 3;
        dst[i] = src[(size_t)h * (NSEQ / 8) + c];
      }
    }
#endif
  };

  stage_k_v(0, 0);

  _Float16* pb = pbuf[wv];
  const _Float16* pr = pb + m * 32 + hl * 16;

  for (int j = 0; j < NSEQ / KB; ++j) {
    const int buf = j & 1;
#if USE_TDM
    if (wv == 0) __builtin_amdgcn_s_wait_tensorcnt(0);
#endif
    __syncthreads();  // tile ready; everyone done reading the other buffer
    if (j + 1 < NSEQ / KB) stage_k_v(buf ^ 1, (j + 1) * KB);

    // scores: S = Q * K^T for 32 keys
    v8f s0 = splat8(0.f), s1 = splat8(0.f);
    const _Float16* k0r = kbuf[buf] + (size_t)m * DIM + hl * 16;
    const _Float16* k1r = k0r + 16 * DIM;
#pragma unroll
    for (int t = 0; t < 16; ++t) {
      v16h kf0 = *(const v16h*)(k0r + t * 32);
      v16h kf1 = *(const v16h*)(k1r + t * 32);
      s0 = wmma_f16(qfr[t], kf0, s0);
      s1 = wmma_f16(qfr[t], kf1, s1);
    }

    // P = exp(S/sqrt(H)); row-sum partials; re-stripe D-layout -> A-layout.
#pragma unroll
    for (int i = 0; i < 8; ++i) {
      float p0 = exp2f(s0[i] * kExpScale);
      float p1 = exp2f(s1[i] * kExpScale);
      lsum[i] += p0 + p1;
      int row = i + 8 * hl;
      pb[row * 32 + m] = (_Float16)p0;
      pb[row * 32 + 16 + m] = (_Float16)p1;
    }
    v8h plo = *(const v8h*)(pr);
    v8h phi = *(const v8h*)(pr + 8);
    v16h pa = cat8(plo, phi);

    // ctx += P * V  (V^T fragments from LDS)
    const _Float16* vr = vbuf[buf] + (size_t)m * KB + hl * 16;
#pragma unroll
    for (int ht = 0; ht < 32; ++ht) {
      v16h vf = *(const v16h*)(vr + ht * 16 * KB);
      ctx[ht] = wmma_f16(pa, vf, ctx[ht]);
    }
  }

  // Row-sum reduction across the 16 lanes of each half, then normalize.
#pragma unroll
  for (int i = 0; i < 8; ++i) {
    float s = lsum[i];
    s += __shfl_xor(s, 1, 32);
    s += __shfl_xor(s, 2, 32);
    s += __shfl_xor(s, 4, 32);
    s += __shfl_xor(s, 8, 32);
    lsum[i] = 1.0f / s;
  }

  _Float16* crow = Ctx + (size_t)rowbase * DIM;
#pragma unroll
  for (int ht = 0; ht < 32; ++ht) {
#pragma unroll
    for (int i = 0; i < 8; ++i)
      crow[(size_t)(i + 8 * hl) * DIM + ht * 16 + m] =
          (_Float16)(ctx[ht][i] * lsum[i]);
  }
#undef STAGE_TILES
}

// ---------------------------------------------------------------------------
// Kernel 3: out = ctx@Wo + bo + X, then LayerNorm.  One wave = 16 full rows.
// ---------------------------------------------------------------------------
__global__ __launch_bounds__(128) void out_ln_kernel(
    const _Float16* __restrict__ Ctx, const _Float16* __restrict__ WoT,
    const float* __restrict__ bo, const float* __restrict__ X,
    const float* __restrict__ gamma, const float* __restrict__ beta,
    float* __restrict__ Out) {
  const int lane = threadIdx.x & 31;
  const int wv = threadIdx.x >> 5;
  const int tile = blockIdx.x * 4 + wv;
  const int rowbase = tile * 16;
  const int m = lane & 15;
  const int hl = lane >> 4;

  v8f acc[32];
#pragma unroll
  for (int ht = 0; ht < 32; ++ht) acc[ht] = splat8(bo[ht * 16 + m]);

  const _Float16* ar = Ctx + (size_t)(rowbase + m) * DIM + hl * 16;
  const _Float16* wr = WoT + (size_t)m * DIM + hl * 16;
  for (int t = 0; t < 16; ++t) {
    v8h alo = *(const v8h*)(ar + t * 32);
    v8h ahi = *(const v8h*)(ar + t * 32 + 8);
    v16h af = cat8(alo, ahi);
#pragma unroll
    for (int ht = 0; ht < 32; ++ht) {
      v16h bf = *(const v16h*)(wr + (size_t)ht * 16 * DIM + t * 32);
      acc[ht] = wmma_f16(af, bf, acc[ht]);
    }
  }

  // Residual add + per-row stats (each lane holds 32 of the 512 columns/row).
  float sum[8], sq[8];
#pragma unroll
  for (int i = 0; i < 8; ++i) { sum[i] = 0.f; sq[i] = 0.f; }
#pragma unroll
  for (int ht = 0; ht < 32; ++ht) {
#pragma unroll
    for (int i = 0; i < 8; ++i) {
      float r = acc[ht][i] +
                X[(size_t)(rowbase + i + 8 * hl) * DIM + ht * 16 + m];
      acc[ht][i] = r;
      sum[i] += r;
      sq[i] += r * r;
    }
  }
#pragma unroll
  for (int i = 0; i < 8; ++i) {
    float s = sum[i], q2 = sq[i];
    s += __shfl_xor(s, 1, 32);  q2 += __shfl_xor(q2, 1, 32);
    s += __shfl_xor(s, 2, 32);  q2 += __shfl_xor(q2, 2, 32);
    s += __shfl_xor(s, 4, 32);  q2 += __shfl_xor(q2, 4, 32);
    s += __shfl_xor(s, 8, 32);  q2 += __shfl_xor(q2, 8, 32);
    float mu = s * (1.0f / DIM);
    float var = q2 * (1.0f / DIM) - mu * mu;
    sum[i] = mu;
    sq[i] = rsqrtf(var + 1e-5f);
  }
#pragma unroll
  for (int ht = 0; ht < 32; ++ht) {
    float g = gamma[ht * 16 + m];
    float be = beta[ht * 16 + m];
#pragma unroll
    for (int i = 0; i < 8; ++i)
      Out[(size_t)(rowbase + i + 8 * hl) * DIM + ht * 16 + m] =
          (acc[ht][i] - sum[i]) * sq[i] * g + be;
  }
}

// ---------------------------------------------------------------------------
extern "C" void kernel_launch(void* const* d_in, const int* in_sizes, int n_in,
                              void* d_out, int out_size, void* d_ws,
                              size_t ws_size, hipStream_t stream) {
  (void)in_sizes; (void)n_in; (void)out_size; (void)ws_size;
  const float* X  = (const float*)d_in[0];
  const float* Wq = (const float*)d_in[1];
  const float* bq = (const float*)d_in[2];
  const float* Wk = (const float*)d_in[3];
  const float* bk = (const float*)d_in[4];
  const float* Wv = (const float*)d_in[5];
  const float* bv = (const float*)d_in[6];
  const float* Wo = (const float*)d_in[7];
  const float* bo = (const float*)d_in[8];
  const float* ln_gamma = (const float*)d_in[9];
  const float* ln_beta  = (const float*)d_in[10];

  const size_t WSZ = (size_t)DIM * DIM * sizeof(_Float16);           // 512 KB
  const size_t MSZ = (size_t)BATCH * NSEQ * DIM * sizeof(_Float16);  // 16 MB
  char* w = (char*)d_ws;
  _Float16* WqT = (_Float16*)(w + 0 * WSZ);
  _Float16* WkT = (_Float16*)(w + 1 * WSZ);
  _Float16* WvT = (_Float16*)(w + 2 * WSZ);
  _Float16* WoT = (_Float16*)(w + 3 * WSZ);
  _Float16* Qh  = (_Float16*)(w + 4 * WSZ);
  _Float16* Kh  = (_Float16*)(w + 4 * WSZ + 1 * MSZ);
  _Float16* Vt  = (_Float16*)(w + 4 * WSZ + 2 * MSZ);
  _Float16* Ctx = (_Float16*)(w + 4 * WSZ + 3 * MSZ);

  const int tiles = BATCH * NSEQ / 16;  // 1024 16-row tiles

  wt_kernel<<<(DIM * DIM) / 256, 256, 0, stream>>>(Wq, Wk, Wv, Wo, WqT, WkT,
                                                   WvT, WoT);
  qkv_kernel<<<tiles / 4, 128, 0, stream>>>(X, WqT, WkT, WvT, bq, bk, bv, Qh,
                                            Kh, Vt);
  attn_kernel<<<tiles / 16, 128, 0, stream>>>(Qh, Kh, Vt, Ctx);
  out_ln_kernel<<<tiles / 4, 128, 0, stream>>>(Ctx, WoT, bo, X, ln_gamma,
                                               ln_beta, (float*)d_out);
}
